// CorrelationHead_24137716203845
// MI455X (gfx1250) — matrix operand order repaired and checked by reading, not compiled, not run on hardware
//
#include <hip/hip_runtime.h>
#include <hip/hip_bf16.h>
#include <stdint.h>

// ---------------------------------------------------------------------------
// Problem constants (from reference)
// ---------------------------------------------------------------------------
#define B_ROIS  1024
#define CCH     256        // channels
#define HH      7
#define SS      49         // 7*7
#define PP      16         // patch size
#define NOUT    12544      // PP*PP*SS  == FC1 K
#define REP     1024       // representation size

typedef __bf16 bf16_t;
typedef __attribute__((ext_vector_type(16))) __bf16 v16bf;
typedef __attribute__((ext_vector_type(8)))  __bf16 v8bf;
typedef __attribute__((ext_vector_type(8)))  float  v8f;

// ---------------------------------------------------------------------------
// CDNA5 async global->LDS copy (16 bytes per lane), tracked by ASYNCcnt.
// LDS address = low 32 bits of the flat shared-aperture address.
// ---------------------------------------------------------------------------
__device__ __forceinline__ void async_copy_b128(const float* g, float* l) {
  unsigned lds_off = (unsigned)(uintptr_t)l;
  asm volatile("global_load_async_to_lds_b128 %0, %1, off"
               :: "v"(lds_off), "v"(g) : "memory");
}
__device__ __forceinline__ void wait_asynccnt0() {
  asm volatile("s_wait_asynccnt 0x0" ::: "memory");
}

// ---------------------------------------------------------------------------
// fp32 -> bf16 conversion (weights)
// ---------------------------------------------------------------------------
__global__ __launch_bounds__(256) void cvt_f32_bf16_kernel(
    const float* __restrict__ in, bf16_t* __restrict__ out, int n)
{
  int i = blockIdx.x * 256 + threadIdx.x;
  if (i < n) out[i] = (bf16_t)in[i];
}

// ---------------------------------------------------------------------------
// Correlation: one block per RoI b.
// corr[b, (ph*16+pw)*49 + i*7 + j] =
//     sum_c x1[b,c,i,j] * x2[b,c,i+2*ph-14, j+2*pw-14]   (0 out of bounds)
// Channel-chunked LDS staging (2 chunks of 128 channels) via async loads,
// 49 outputs/thread in registers. Output bf16 row-major [B, 12544].
// ---------------------------------------------------------------------------
__global__ __launch_bounds__(256) void corr_kernel(
    const float* __restrict__ x1, const float* __restrict__ x2,
    bf16_t* __restrict__ corr)
{
  __shared__ float s1[128 * SS];
  __shared__ float s2[128 * SS];

  const int b   = blockIdx.x;
  const int tid = threadIdx.x;
  const float* p1 = x1 + (size_t)b * CCH * SS;
  const float* p2 = x2 + (size_t)b * CCH * SS;

  float acc[49];
  #pragma unroll
  for (int oi = 0; oi < 49; ++oi) acc[oi] = 0.0f;

  for (int cb = 0; cb < 2; ++cb) {
    __syncthreads();              // previous chunk fully consumed
    const int base = cb * 128 * SS;
    const int nvec = (128 * SS) / 4;          // 1568 16-byte chunks
    for (int i = tid; i < nvec; i += 256) {
      async_copy_b128(p1 + base + i * 4, &s1[i * 4]);
      async_copy_b128(p2 + base + i * 4, &s2[i * 4]);
    }
    wait_asynccnt0();
    __syncthreads();

    for (int oi = 0; oi < 49; ++oi) {
      const int idx = oi * 256 + tid;     // 0 .. 12543 (exactly covered)
      const int p = idx / SS;             // displacement index 0..255
      const int s = idx - p * SS;         // spatial 0..48
      const int i0 = s / HH, j0 = s - i0 * HH;
      const int i2 = i0 + 2 * (p >> 4) - 14;
      const int j2 = j0 + 2 * (p & 15) - 14;
      if ((unsigned)i2 < (unsigned)HH && (unsigned)j2 < (unsigned)HH) {
        const int s2i = i2 * HH + j2;
        float a = acc[oi];
        #pragma unroll 8
        for (int c = 0; c < 128; ++c)
          a = fmaf(s1[c * SS + s], s2[c * SS + s2i], a);
        acc[oi] = a;
      }
    }
  }

  bf16_t* o = corr + (size_t)b * NOUT;
  #pragma unroll
  for (int oi = 0; oi < 49; ++oi)
    o[oi * 256 + tid] = (bf16_t)acc[oi];
}

// ---------------------------------------------------------------------------
// WMMA helpers: CDNA5 16-bit fragment striping.
//  A (16x32, MxK): lane L (L<16) = row M=L, K in {k..k+7, k+16..k+23};
//                  lane L+16     = row M=L, K in {k+8..k+15, k+24..k+31}.
//  B (32x16, KxN): lane L (L<16) = col N=L, K in {k..k+7, k+16..k+23};
//                  lane L+16     = col N=L, K in {k+8..k+15, k+24..k+31}.
//  C/D: VGPR r -> row mtile + r + 8*(lane>=16), col ntile + (lane&15).
// ---------------------------------------------------------------------------
__device__ __forceinline__ v16bf load_a_frag(const bf16_t* p) {
  v8bf lo = *(const v8bf*)(p);        // K = k + half*8      .. +7
  v8bf hi = *(const v8bf*)(p + 16);   // K = k + 16 + half*8 .. +7
  v16bf a;
  #pragma unroll
  for (int i = 0; i < 8; ++i) { a[i] = lo[i]; a[8 + i] = hi[i]; }
  return a;
}
__device__ __forceinline__ v8f wmma_bf16(v16bf a, v16bf b, v8f c) {
  return __builtin_amdgcn_wmma_f32_16x16x32_bf16(false, a, false, b,
                                                 (short)0, c, false, false);
}

// ---------------------------------------------------------------------------
// GEMM:  out[M,N] = act(A[M,K] * W[N,K]^T + bias)   (bf16 in, bf16 out)
// One wave computes a 32(M) x 64(N) register tile: 2 A-frags x 4 B-frags ->
// 8 WMMAs per K-step of 32. 8 waves/block. Requires M%32==0, N%64==0.
// ---------------------------------------------------------------------------
__global__ __launch_bounds__(256) void gemm_bf16_kernel(
    const bf16_t* __restrict__ A,     // [M, K] row-major
    const bf16_t* __restrict__ W,     // [N, K] row-major
    const float*  __restrict__ bias,  // [N]
    bf16_t* __restrict__ out,         // [M, N]
    int M, int N, int K, int relu)
{
  const int lane   = threadIdx.x & 31;
  const int wave   = threadIdx.x >> 5;
  const int ntile0 = (blockIdx.x * 8 + wave) * 64;
  const int mtile0 = blockIdx.y * 32;
  if (ntile0 >= N || mtile0 >= M) return;

  const int half = lane >> 4;   // 0: lanes 0-15, 1: lanes 16-31
  const int l15  = lane & 15;

  const bf16_t* Arow0 = A + (size_t)(mtile0 + l15) * K + half * 8;
  const bf16_t* Arow1 = Arow0 + (size_t)16 * K;
  const bf16_t* Wrow  = W + (size_t)(ntile0 + l15) * K + half * 8;
  const size_t  wstep = (size_t)16 * K;   // 16 output columns

  v8f zero = {};
  v8f acc[8];
  #pragma unroll
  for (int i = 0; i < 8; ++i) acc[i] = zero;

  #pragma unroll 2
  for (int k = 0; k < K; k += 32) {
    v16bf a0 = load_a_frag(Arow0 + k);
    v16bf a1 = load_a_frag(Arow1 + k);
    #pragma unroll
    for (int nt = 0; nt < 4; ++nt) {
      v16bf bm = load_a_frag(Wrow + (size_t)nt * wstep + k);
      acc[nt]     = wmma_bf16(a0, bm, acc[nt]);
      acc[4 + nt] = wmma_bf16(a1, bm, acc[4 + nt]);
    }
  }

  #pragma unroll
  for (int nt = 0; nt < 4; ++nt) {
    const int   col = ntile0 + nt * 16 + l15;
    const float bv  = bias[col];
    #pragma unroll
    for (int r = 0; r < 8; ++r) {
      const int row0 = mtile0 + r + half * 8;        // A-frag 0 rows
      const int row1 = row0 + 16;                    // A-frag 1 rows
      float v0 = acc[nt][r] + bv;
      float v1 = acc[4 + nt][r] + bv;
      if (relu) { v0 = v0 > 0.0f ? v0 : 0.0f; v1 = v1 > 0.0f ? v1 : 0.0f; }
      out[(size_t)row0 * N + col] = (bf16_t)v0;
      out[(size_t)row1 * N + col] = (bf16_t)v1;
    }
  }
}

// ---------------------------------------------------------------------------
// FC3: out[1024,4] = fc2[1024,1024] * W3[4,1024]^T + b3   (fp32 out)
// Tiny N: plain VALU dot products.
// ---------------------------------------------------------------------------
__global__ __launch_bounds__(128) void fc3_kernel(
    const bf16_t* __restrict__ x, const bf16_t* __restrict__ W3,
    const float* __restrict__ b3, float* __restrict__ out)
{
  const int t = blockIdx.x * 128 + threadIdx.x;   // 0..4095
  const int m = t >> 2, n = t & 3;
  const bf16_t* xr = x + (size_t)m * REP;
  const bf16_t* wr = W3 + (size_t)n * REP;
  float acc = 0.0f;
  #pragma unroll 8
  for (int k = 0; k < REP; ++k)
    acc = fmaf((float)xr[k], (float)wr[k], acc);
  out[t] = acc + b3[n];
}

// ---------------------------------------------------------------------------
// Launch
// ---------------------------------------------------------------------------
extern "C" void kernel_launch(void* const* d_in, const int* in_sizes, int n_in,
                              void* d_out, int out_size, void* d_ws, size_t ws_size,
                              hipStream_t stream) {
  const float* patch1 = (const float*)d_in[0];   // [1024,256,7,7]
  const float* patch2 = (const float*)d_in[1];   // [1024,256,7,7]
  const float* W1     = (const float*)d_in[2];   // [1024,12544]
  const float* b1     = (const float*)d_in[3];   // [1024]
  const float* W2     = (const float*)d_in[4];   // [1024,1024]
  const float* b2     = (const float*)d_in[5];   // [1024]
  const float* W3     = (const float*)d_in[6];   // [4,1024]
  const float* b3     = (const float*)d_in[7];   // [4]
  float* out = (float*)d_out;                    // [1024,4]

  // Workspace carve-up (all offsets 256B aligned)
  uint8_t* ws = (uint8_t*)d_ws;
  const size_t corr_bytes = (size_t)B_ROIS * NOUT * 2;   // 25,690,112
  const size_t w1b_bytes  = (size_t)REP * NOUT * 2;      // 25,690,112
  const size_t w2b_bytes  = (size_t)REP * REP * 2;       //  2,097,152
  const size_t w3b_bytes  = (size_t)4 * REP * 2;         //      8,192
  const size_t fcb_bytes  = (size_t)B_ROIS * REP * 2;    //  2,097,152

  bf16_t* corr = (bf16_t*)(ws);
  bf16_t* W1b  = (bf16_t*)(ws + corr_bytes);
  bf16_t* W2b  = (bf16_t*)(ws + corr_bytes + w1b_bytes);
  bf16_t* W3b  = (bf16_t*)(ws + corr_bytes + w1b_bytes + w2b_bytes);
  bf16_t* fc1  = (bf16_t*)(ws + corr_bytes + w1b_bytes + w2b_bytes + w3b_bytes);
  bf16_t* fc2  = (bf16_t*)(ws + corr_bytes + w1b_bytes + w2b_bytes + w3b_bytes + fcb_bytes);

  // 1. Weight conversion fp32 -> bf16
  {
    int n = REP * NOUT;
    cvt_f32_bf16_kernel<<<(n + 255) / 256, 256, 0, stream>>>(W1, W1b, n);
  }
  {
    int n = REP * REP;
    cvt_f32_bf16_kernel<<<(n + 255) / 256, 256, 0, stream>>>(W2, W2b, n);
  }
  {
    int n = 4 * REP;
    cvt_f32_bf16_kernel<<<(n + 255) / 256, 256, 0, stream>>>(W3, W3b, n);
  }

  // 2. Correlation -> bf16 activation matrix [1024, 12544]
  corr_kernel<<<B_ROIS, 256, 0, stream>>>(patch1, patch2, corr);

  // 3. FC1: [1024,12544] x [1024,12544]^T -> relu -> bf16 [1024,1024]
  {
    dim3 grid(REP / 64 / 8, B_ROIS / 32);   // (2, 32)
    gemm_bf16_kernel<<<grid, 256, 0, stream>>>(corr, W1b, b1, fc1,
                                               B_ROIS, REP, NOUT, 1);
  }

  // 4. FC2: [1024,1024] x [1024,1024]^T -> relu -> bf16 [1024,1024]
  {
    dim3 grid(REP / 64 / 8, B_ROIS / 32);   // (2, 32)
    gemm_bf16_kernel<<<grid, 256, 0, stream>>>(fc1, W2b, b2, fc2,
                                               B_ROIS, REP, REP, 1);
  }

  // 5. FC3: [1024,1024] x [4,1024]^T + b3 -> fp32 [1024,4]
  fc3_kernel<<<(B_ROIS * 4) / 128, 128, 0, stream>>>(fc2, W3b, b3, out);
}
